// ClusterTransformerBlock_3375844295245
// MI455X (gfx1250) — matrix operand order repaired and checked by reading, not compiled, not run on hardware
//
#include <hip/hip_runtime.h>
#include <hip/hip_bf16.h>

// ---------------- problem constants ----------------
constexpr int BSZ = 2, NN = 8192, CC = 192, HH = 6, MM = 48, TT = 3025, HID = 384;
constexpr int ROWS = BSZ * NN;           // 16384
constexpr int KSTR = 196;                // padded LDS row stride (floats): 784B = 16B-aligned, 196%64=4

typedef __attribute__((ext_vector_type(16))) __bf16 v16bf;
typedef __attribute__((ext_vector_type(8)))  float  v8f;

__device__ __forceinline__ unsigned short f2bf(float f) {
    unsigned int u = __builtin_bit_cast(unsigned int, f);
    u += 0x7fffu + ((u >> 16) & 1u);     // round-to-nearest-even
    return (unsigned short)(u >> 16);
}
__device__ __forceinline__ __bf16 us2bf(unsigned short u) {
    return __builtin_bit_cast(__bf16, u);
}
__device__ __forceinline__ float gelu_exact(float x) {
    return 0.5f * x * (1.0f + erff(x * 0.70710678118654752440f));
}

// ---------------- CDNA5 async global->LDS DMA helpers ----------------
__device__ __forceinline__ unsigned lds_addr(const void* p) {
    // generic pointers to LDS carry the LDS byte address in the low 32 bits
    return (unsigned)(unsigned long long)p;
}
__device__ __forceinline__ void async_load_b128(unsigned lds_off, const void* gaddr) {
    asm volatile("global_load_async_to_lds_b128 %0, %1, off"
                 :: "v"(lds_off), "v"(gaddr) : "memory");
}
__device__ __forceinline__ void async_load_b32(unsigned lds_off, const void* gaddr) {
    asm volatile("global_load_async_to_lds_b32 %0, %1, off"
                 :: "v"(lds_off), "v"(gaddr) : "memory");
}
__device__ __forceinline__ void wait_async0() {
    asm volatile("s_wait_asynccnt 0x0" ::: "memory");
}

// ---------------- kernel 1: weight conversion (-> bf16, TRANSPOSED) + PE table ----------------
__global__ __launch_bounds__(256) void prep_kernel(
    const float* __restrict__ q_w, const float* __restrict__ kv_w,
    const float* __restrict__ proj_w, const float* __restrict__ fc1_w,
    const float* __restrict__ fc2_w, const float* __restrict__ pre_table,
    const float* __restrict__ pe_w, const float* __restrict__ pe_b,
    unsigned short* __restrict__ wqkvT, unsigned short* __restrict__ wprojT,
    unsigned short* __restrict__ wfc1T, unsigned short* __restrict__ wfc2T,
    float* __restrict__ pe_tab)
{
    int i = blockIdx.x * 256 + threadIdx.x;
    const int nqkv = CC * (3 * CC);      // 110592
    const int nproj = CC * CC;           // 36864
    const int nfc1 = CC * HID;           // 73728
    const int nfc2 = HID * CC;           // 73728
    const int npe  = TT * HH;            // 18150
    if (i < nqkv) {
        int k = i / (3 * CC), n = i % (3 * CC);
        float v = (n < CC) ? q_w[k * CC + n] : kv_w[k * (2 * CC) + (n - CC)];
        wqkvT[(size_t)n * CC + k] = f2bf(v);              // (576 x 192)
        return;
    }
    i -= nqkv;
    if (i < nproj) {
        int k = i / CC, n = i % CC;
        wprojT[(size_t)n * CC + k] = f2bf(proj_w[i]);      // (192 x 192)
        return;
    }
    i -= nproj;
    if (i < nfc1) {
        int k = i / HID, n = i % HID;
        wfc1T[(size_t)n * CC + k] = f2bf(fc1_w[i]);        // (384 x 192)
        return;
    }
    i -= nfc1;
    if (i < nfc2) {
        int k = i / CC, n = i % CC;
        wfc2T[(size_t)n * HID + k] = f2bf(fc2_w[i]);       // (192 x 384)
        return;
    }
    i -= nfc2;
    if (i < npe) {
        int t = i / HH, h = i % HH;
        float acc = pe_b[h];
        #pragma unroll
        for (int p = 0; p < 5; ++p) acc += pre_table[t * 5 + p] * pe_w[p * HH + h];
        pe_tab[i] = acc;
    }
}

// ---------------- kernel 2: LayerNorm (fp32 in) -> bf16 out ----------------
// one wave32 per row of 192; 8 rows per 256-thread block
__global__ __launch_bounds__(256) void ln_bf16_kernel(
    const float* __restrict__ x, const float* __restrict__ g,
    const float* __restrict__ b, unsigned short* __restrict__ y)
{
    const int row  = blockIdx.x * 8 + (threadIdx.x >> 5);
    const int lane = threadIdx.x & 31;
    const float* xr = x + (size_t)row * CC;
    float v[6];
    float s = 0.f;
    #pragma unroll
    for (int j = 0; j < 6; ++j) { v[j] = xr[lane + 32 * j]; s += v[j]; }
    #pragma unroll
    for (int off = 16; off > 0; off >>= 1) s += __shfl_xor(s, off, 32);
    const float mu = s * (1.0f / CC);
    float s2 = 0.f;
    #pragma unroll
    for (int j = 0; j < 6; ++j) { float d = v[j] - mu; s2 += d * d; }
    #pragma unroll
    for (int off = 16; off > 0; off >>= 1) s2 += __shfl_xor(s2, off, 32);
    const float rstd = rsqrtf(s2 * (1.0f / CC) + 1e-5f);
    #pragma unroll
    for (int j = 0; j < 6; ++j) {
        int c = lane + 32 * j;
        y[(size_t)row * CC + c] = f2bf((v[j] - mu) * rstd * g[c] + b[c]);
    }
}

// ---------------- WMMA GEMM: (ROWS x K) bf16 @ W^T (NC x K) bf16 -> epilogue ----------------
// Double-buffered async-to-LDS pipeline; both A and B^T tiles land K-contiguous so every
// fragment load is 2x ds_load_b128.
// MODE 0: QKV split-store (q scaled, k/v per head/pair), MODE 1: +bias+resid -> fp32,
// MODE 2: +bias, exact GELU -> bf16
template<int K, int NC, int MODE>
__global__ __launch_bounds__(256) void gemm_bf16_kernel(
    const unsigned short* __restrict__ A, const unsigned short* __restrict__ WT,
    const float* __restrict__ bias0, const float* __restrict__ bias1,
    const float* __restrict__ resid,
    float* __restrict__ outf, float* __restrict__ outk, float* __restrict__ outv,
    unsigned short* __restrict__ outh)
{
    __shared__ __align__(16) unsigned short lA[2][64][40];   // 64 rows x 32 k, padded
    __shared__ __align__(16) unsigned short lB[2][64][40];   // 64 cols x 32 k (transposed), padded
    const int tid  = threadIdx.x;
    const int row0 = blockIdx.x * 64;
    const int col0 = blockIdx.y * 64;
    const int w    = tid >> 5;
    const int lane = tid & 31;
    const int tm0  = w >> 2;                // tiles: (tm0, tn) and (tm0+2, tn) share B fragment
    const int tn   = w & 3;
    const int mrow = lane & 15;
    const int lo   = (lane < 16) ? 0 : 8;   // unused; A-fragment K base is kb
    const int kb   = (lane < 16) ? 0 : 16;  // B-fragment K base
    const int rS   = tid >> 2;              // staging: row/col 0..63
    const int cgS  = (tid & 3) * 8;         // staging: k-offset 0,8,16,24 (16B chunks)
    v8f acc0 = {}; v8f acc1 = {};

    const unsigned short* srcA = A  + (size_t)(row0 + rS) * K + cgS;
    const unsigned short* srcB = WT + (size_t)(col0 + rS) * K + cgS;

    // prologue: stage k-tile 0 into buffer 0
    async_load_b128(lds_addr(&lA[0][rS][cgS]), srcA);
    async_load_b128(lds_addr(&lB[0][rS][cgS]), srcB);

    int buf = 0;
    for (int k0 = 0; k0 < K; k0 += 32) {
        wait_async0();
        __syncthreads();
        if (k0 + 32 < K) {   // issue next tile into the other buffer while computing
            async_load_b128(lds_addr(&lA[buf ^ 1][rS][cgS]), srcA + k0 + 32);
            async_load_b128(lds_addr(&lB[buf ^ 1][rS][cgS]), srcB + k0 + 32);
        }
        v16bf bfrag;
        #pragma unroll
        for (int e = 0; e < 16; ++e) bfrag[e] = us2bf(lB[buf][tn * 16 + mrow][kb + e]);
        v16bf a0, a1;
        #pragma unroll
        for (int e = 0; e < 16; ++e) {
            int kk = kb + e;
            a0[e] = us2bf(lA[buf][tm0 * 16 + mrow][kk]);
            a1[e] = us2bf(lA[buf][(tm0 + 2) * 16 + mrow][kk]);
        }
        acc0 = __builtin_amdgcn_wmma_f32_16x16x32_bf16(false, a0, false, bfrag,
                                                       (short)0, acc0, false, false);
        acc1 = __builtin_amdgcn_wmma_f32_16x16x32_bf16(false, a1, false, bfrag,
                                                       (short)0, acc1, false, false);
        buf ^= 1;
    }

    const int col = col0 + tn * 16 + mrow;
    #pragma unroll
    for (int e = 0; e < 8; ++e) {
        int r0 = row0 + tm0 * 16 + e + ((lane < 16) ? 0 : 8);
        int r1 = r0 + 32;
        float vals[2] = { acc0[e], acc1[e] };
        int   rows[2] = { r0, r1 };
        #pragma unroll
        for (int q2 = 0; q2 < 2; ++q2) {
            float v = vals[q2];
            int   r = rows[q2];
            if constexpr (MODE == 0) {
                if (col < CC) {
                    outf[(size_t)r * CC + col] = (v + bias0[col]) * 0.17677669529663688f; // 32^-0.5
                } else {
                    int c2 = col - CC;                 // [0,384): h*64 + pair*32 + c_
                    float vv = v + bias1[c2];
                    int h = c2 >> 6, rem = c2 & 63;
                    size_t dst = (size_t)r * CC + h * 32 + (rem & 31);
                    if (rem < 32) outk[dst] = vv; else outv[dst] = vv;
                }
            } else if constexpr (MODE == 1) {
                size_t d = (size_t)r * NC + col;
                outf[d] = v + bias0[col] + resid[d];
            } else {  // MODE 2
                outh[(size_t)r * NC + col] = f2bf(gelu_exact(v + bias0[col]));
            }
        }
    }
}

// ---------------- attention kernel: one 192-thread block per query ----------------
__global__ __launch_bounds__(192) void attn_kernel(
    const float* __restrict__ q, const float* __restrict__ k, const float* __restrict__ v,
    const int* __restrict__ member_idx, const int* __restrict__ cluster_mask,
    const int* __restrict__ pe_idx, const float* __restrict__ pe_tab,
    const float* __restrict__ blank_k, const float* __restrict__ blank_v,
    unsigned short* __restrict__ out_bf)
{
    extern __shared__ float sm[];
    float* kbuf = sm;                       // MM * KSTR
    float* vbuf = kbuf + MM * KSTR;         // MM * KSTR
    float* qbuf = vbuf + MM * KSTR;         // CC
    float* sc   = qbuf + CC;                // HH * 64 (49 used per head)
    const int t = threadIdx.x;
    const size_t row = (size_t)blockIdx.y * NN + blockIdx.x;

    qbuf[t] = q[row * CC + t];
    // async DMA gather: per iteration 192 lanes move 16B each = 2 K-rows + 2 V-rows
    {
        const int sub = t / 96;             // which of the two m's this lane serves
        const int r   = t % 96;             // 0..47 -> K row chunk, 48..95 -> V row chunk
        const int c   = (r < 48) ? r : (r - 48);
        const float* base = (r < 48) ? k : v;
        float* dbase      = (r < 48) ? kbuf : vbuf;
        for (int mm = 0; mm < MM; mm += 2) {
            int m = mm + sub;
            int ridx = member_idx[row * MM + m];
            const float* src = base + ((size_t)blockIdx.y * NN + ridx) * CC + c * 4;
            async_load_b128(lds_addr(dbase + m * KSTR + c * 4), src);
        }
    }
    wait_async0();
    __syncthreads();

    // scores: 6 heads x 48 neighbors
    for (int s = t; s < HH * MM; s += 192) {
        int h = s / MM, m = s % MM;
        const float* qh = qbuf + h * 32;
        const float* kr = kbuf + m * KSTR + h * 32;
        float acc = 0.f;
        #pragma unroll
        for (int i = 0; i < 32; ++i) acc += qh[i] * kr[i];
        acc += pe_tab[(size_t)pe_idx[row * MM + m] * HH + h];
        acc += (1.0f - (float)cluster_mask[row * MM + m]) * -100.0f;
        sc[h * 64 + m] = acc;
    }
    if (t < HH) {   // blank logit
        const float* qh = qbuf + t * 32;
        const float* bk = blank_k + t * 32;
        float acc = 0.f;
        #pragma unroll
        for (int i = 0; i < 32; ++i) acc += qh[i] * bk[i];
        sc[t * 64 + MM] = acc;
    }
    __syncthreads();

    if (t < HH) {   // softmax over 49 per head
        float mx = -1e30f;
        for (int i = 0; i <= MM; ++i) mx = fmaxf(mx, sc[t * 64 + i]);
        float sum = 0.f;
        for (int i = 0; i <= MM; ++i) { float e = __expf(sc[t * 64 + i] - mx); sc[t * 64 + i] = e; sum += e; }
        float inv = 1.0f / sum;
        for (int i = 0; i <= MM; ++i) sc[t * 64 + i] *= inv;
    }
    __syncthreads();

    {   // out[c] = sum_m p[h][m] * v[m][c] + p_blank[h] * blank_v[c]
        int h = t >> 5;
        float acc = sc[h * 64 + MM] * blank_v[t];
        for (int m = 0; m < MM; ++m) acc += sc[h * 64 + m] * vbuf[m * KSTR + t];
        out_bf[row * CC + t] = f2bf(acc);
    }
}

// ---------------- launcher ----------------
extern "C" void kernel_launch(void* const* d_in, const int* in_sizes, int n_in,
                              void* d_out, int out_size, void* d_ws, size_t ws_size,
                              hipStream_t stream) {
    const float* feat        = (const float*)d_in[0];
    const int*   member_idx  = (const int*)d_in[1];
    const int*   cluster_mask= (const int*)d_in[2];
    const int*   pe_idx      = (const int*)d_in[3];
    // d_in[4] = global_attn (0, unused)
    const float* pre_table   = (const float*)d_in[5];
    const float* norm1_g     = (const float*)d_in[6];
    const float* norm1_b     = (const float*)d_in[7];
    const float* q_w         = (const float*)d_in[8];
    const float* q_b         = (const float*)d_in[9];
    const float* kv_w        = (const float*)d_in[10];
    const float* kv_b        = (const float*)d_in[11];
    const float* blank_k     = (const float*)d_in[12];
    const float* blank_v     = (const float*)d_in[13];
    const float* pe_w        = (const float*)d_in[14];
    const float* pe_b        = (const float*)d_in[15];
    const float* proj_w      = (const float*)d_in[16];
    const float* proj_b      = (const float*)d_in[17];
    const float* norm2_g     = (const float*)d_in[18];
    const float* norm2_b     = (const float*)d_in[19];
    const float* fc1_w       = (const float*)d_in[20];
    const float* fc1_b       = (const float*)d_in[21];
    const float* fc2_w       = (const float*)d_in[22];
    const float* fc2_b       = (const float*)d_in[23];
    float* out = (float*)d_out;

    char* ws = (char*)d_ws;
    size_t off = 0;
    auto take = [&](size_t bytes) -> void* {
        void* p = ws + off;
        off = (off + bytes + 255) & ~(size_t)255;
        return p;
    };
    unsigned short* wqkvT = (unsigned short*)take((size_t)CC * 3 * CC * 2);
    unsigned short* wprojT= (unsigned short*)take((size_t)CC * CC * 2);
    unsigned short* wfc1T = (unsigned short*)take((size_t)CC * HID * 2);
    unsigned short* wfc2T = (unsigned short*)take((size_t)HID * CC * 2);
    float*          petab = (float*)take((size_t)TT * HH * 4);
    unsigned short* xbf   = (unsigned short*)take((size_t)ROWS * CC * 2);   // LN1 out, reused for LN2 out
    float*          qbuf  = (float*)take((size_t)ROWS * CC * 4);
    float*          kbuf  = (float*)take((size_t)ROWS * CC * 4);
    float*          vbuf  = (float*)take((size_t)ROWS * CC * 4);
    unsigned short* abf   = (unsigned short*)take((size_t)ROWS * CC * 2);   // attention out (bf16)
    float*          feat2 = (float*)take((size_t)ROWS * CC * 4);
    unsigned short* hbf   = (unsigned short*)take((size_t)ROWS * HID * 2);  // gelu(fc1) out

    // 1) weights -> bf16 transposed, PE table
    {
        int total = CC * 3 * CC + CC * CC + CC * HID + HID * CC + TT * HH;
        prep_kernel<<<(total + 255) / 256, 256, 0, stream>>>(
            q_w, kv_w, proj_w, fc1_w, fc2_w, pre_table, pe_w, pe_b,
            wqkvT, wprojT, wfc1T, wfc2T, petab);
    }
    // 2) LN1 -> bf16
    ln_bf16_kernel<<<ROWS / 8, 256, 0, stream>>>(feat, norm1_g, norm1_b, xbf);
    // 3) fused QKV GEMM (WMMA) with scale/split epilogue
    gemm_bf16_kernel<CC, 3 * CC, 0><<<dim3(ROWS / 64, (3 * CC) / 64), 256, 0, stream>>>(
        xbf, wqkvT, q_b, kv_b, nullptr, qbuf, kbuf, vbuf, nullptr);
    // 4) gather attention (L2-resident K/V, async-DMA staged per query)
    {
        size_t smem = (size_t)(2 * MM * KSTR + CC + HH * 64) * sizeof(float);
        attn_kernel<<<dim3(NN, BSZ), 192, smem, stream>>>(
            qbuf, kbuf, vbuf, member_idx, cluster_mask, pe_idx, petab,
            blank_k, blank_v, abf);
    }
    // 5) proj GEMM + residual -> feat2
    gemm_bf16_kernel<CC, CC, 1><<<dim3(ROWS / 64, CC / 64), 256, 0, stream>>>(
        abf, wprojT, proj_b, nullptr, feat, feat2, nullptr, nullptr, nullptr);
    // 6) LN2 -> bf16 (reuse xbf)
    ln_bf16_kernel<<<ROWS / 8, 256, 0, stream>>>(feat2, norm2_g, norm2_b, xbf);
    // 7) FC1 GEMM + exact GELU -> bf16
    gemm_bf16_kernel<CC, HID, 2><<<dim3(ROWS / 64, HID / 64), 256, 0, stream>>>(
        xbf, wfc1T, fc1_b, nullptr, nullptr, nullptr, nullptr, nullptr, hbf);
    // 8) FC2 GEMM + bias + feat2 residual -> final output
    gemm_bf16_kernel<HID, CC, 1><<<dim3(ROWS / 64, CC / 64), 256, 0, stream>>>(
        hbf, wfc2T, fc2_b, nullptr, feat2, out, nullptr, nullptr, nullptr);
}
